// encoder_17540646437531
// MI455X (gfx1250) — compile-verified, matrix operand order checked by reading, not run on hardware
//
#include <hip/hip_runtime.h>
#include <hip/hip_bf16.h>

// ---------------------------------------------------------------------------
// Problem constants (from the reference)
// ---------------------------------------------------------------------------
#define BB    2048
#define TT    128
#define HH    128
#define NN    256            // CTX(64) + SEMB(32) + UEMB(32) + BASE(128)
#define GG    512            // 4*H gate width
#define KK    384            // N + H  (concatenated GEMM K)
#define KCH   12             // KK / 32 WMMA K-chunks (0..7 = wx, 8..11 = h)
#define ROWS  16             // batch rows per workgroup (one WMMA M tile)
#define HP    136            // padded LDS pitch (halves) for h state
#define WXP   264            // padded LDS pitch (halves) for staged wx tile

typedef __attribute__((ext_vector_type(16))) __bf16 v16bf;
typedef __attribute__((ext_vector_type(8)))  float  v8f;
typedef unsigned short u16;

union V16 {
  v16bf v;
  u16   u[16];
  uint4 q[2];
};

// fp32 -> bf16 round-to-nearest-even (bit manipulation)
static __device__ __forceinline__ u16 f2bf(float f) {
  unsigned int u = __float_as_uint(f);
  u += 0x7FFFu + ((u >> 16) & 1u);
  return (u16)(u >> 16);
}

// x[b,t,n] gather: concat(context[64], sport_emb[32], user_emb[32], input_var[128])
static __device__ __forceinline__ float gather_x(int n, size_t bt,
    const float* __restrict__ ctx, const float* __restrict__ iv,
    const float* __restrict__ utab, const float* __restrict__ stab,
    int si, int ui) {
  if (n < 64)        return ctx[bt * 64 + n];
  else if (n < 96)   return stab[(size_t)si * 32 + (n - 64)];
  else if (n < 128)  return utab[(size_t)ui * 32 + (n - 96)];
  else               return iv[bt * 128 + (n - 128)];
}

// ---------------------------------------------------------------------------
// Kernel 1: pack [W_ih ; W_hh] into bf16 Wcat[512][384] (row-major).
// Row-major rows == WMMA B-matrix columns => per-lane contiguous K loads.
// ---------------------------------------------------------------------------
__global__ void pack_w_kernel(const float* __restrict__ Wih,
                              const float* __restrict__ Whh,
                              u16* __restrict__ Wcat) {
  int i = blockIdx.x * blockDim.x + threadIdx.x;
  if (i >= GG * KK) return;
  int row = i / KK;
  int k   = i - row * KK;
  float v = (k < NN) ? Wih[(size_t)row * NN + k]
                     : Whh[(size_t)row * HH + (k - NN)];
  Wcat[i] = f2bf(v);
}

// ---------------------------------------------------------------------------
// Kernel 2: series_score + softmax (shift-invariant: the per-batch scalar hc
// term of the reference cancels exactly in softmax) + wx = attn * x.
// Writes wx fp32 into d_out; optionally also a bf16 copy into d_ws.
// ---------------------------------------------------------------------------
template <bool WRITE_BF>
__global__ __launch_bounds__(256) void attn_wx_kernel(
    const int*   __restrict__ uin, const int* __restrict__ sin,
    const float* __restrict__ ctx, const float* __restrict__ iv,
    const float* __restrict__ utab, const float* __restrict__ stab,
    const float* __restrict__ attw, float* __restrict__ wx,
    u16* __restrict__ wxbf) {
  const int b = blockIdx.x;
  const int n = threadIdx.x;

  __shared__ float red[256];
  __shared__ int   sidx[TT];
  __shared__ int   uidx[TT];
  __shared__ float wt[TT];

  if (n < TT) {
    sidx[n] = sin[(size_t)b * TT + n];
    uidx[n] = uin[(size_t)b * TT + n];
    wt[n]   = attw[2 * HH + n];          // w_t slice of attn_w
  }
  __syncthreads();

  float ss = 0.0f;
  for (int t = 0; t < TT; ++t) {
    size_t bt = (size_t)b * TT + t;
    float x = gather_x(n, bt, ctx, iv, utab, stab, sidx[t], uidx[t]);
    ss = fmaf(x, wt[t], ss);
  }

  // block softmax over the 256 features
  red[n] = ss; __syncthreads();
  for (int s = 128; s > 0; s >>= 1) {
    if (n < s) red[n] = fmaxf(red[n], red[n + s]);
    __syncthreads();
  }
  float m = red[0]; __syncthreads();
  float e = __expf(ss - m);
  red[n] = e; __syncthreads();
  for (int s = 128; s > 0; s >>= 1) {
    if (n < s) red[n] += red[n + s];
    __syncthreads();
  }
  float a = e / red[0];

  for (int t = 0; t < TT; ++t) {
    size_t bt = (size_t)b * TT + t;
    float x = gather_x(n, bt, ctx, iv, utab, stab, sidx[t], uidx[t]);
    float w = a * x;
    wx[bt * NN + n] = w;                 // input_weighted output (fp32)
    if (WRITE_BF) wxbf[bt * NN + n] = f2bf(w);
  }
}

// ---------------------------------------------------------------------------
// Kernel 3: LSTM recurrence. Block = 16 batch rows, 256 threads = 8 waves.
// Each wave owns 4 of the 32 gate N-tiles. Per step:
//   gates(16x512) = [wx_t(16x256) | h(16x128)] @ Wcat^T via 12x4 bf16 WMMA.
// - All 48 B-tiles are re-loaded from global Wcat inside the t-loop, every step.
// - wx tile for step t+1 is staged through LDS with a software pipeline:
//   global loads issue at the TOP of step t, the ds_store (and thus the
//   loadcnt wait) happens at the BOTTOM, so HBM/L2 latency is hidden behind
//   the whole step. A-chunks then read from LDS at LDS latency.
// - h/c state lives in LDS for all 128 steps; no inter-block sync needed.
// ---------------------------------------------------------------------------
template <bool BF16WX>
__global__ __launch_bounds__(256) void lstm_rec_kernel(
    const float* __restrict__ wx,        // d_out input_weighted region (fp32)
    const u16*   __restrict__ wxbf,      // bf16 copy in d_ws (BF16WX path)
    const u16*   __restrict__ Wcat,      // bf16 [512][384]
    const float* __restrict__ bih, const float* __restrict__ bhh,
    float* __restrict__ enc) {           // d_out input_encoded region
  const int b0   = blockIdx.x * ROWS;
  const int tid  = threadIdx.x;
  const int lane = tid & 31;
  const int wave = tid >> 5;
  const int hi   = lane >> 4;            // half-wave select (K window / D row)
  const int lm   = lane & 15;            // A row / B col / D col in tile

  __shared__ float gates[ROWS][GG + 4];  // +4 pad vs bank conflicts
  __shared__ float c_st[ROWS][HH];
  __shared__ u16   h_bf[ROWS][HP];       // bf16 h state, padded pitch
  __shared__ u16   wxst[2][ROWS * WXP];  // double-buffered staged wx tile
  __shared__ float bg[GG];

  for (int i = tid; i < GG; i += 256)        bg[i] = bih[i] + bhh[i];
  for (int i = tid; i < ROWS * HH; i += 256) ((float*)c_st)[i] = 0.0f;
  for (int i = tid; i < ROWS * HP; i += 256) ((u16*)h_bf)[i]   = (u16)0;

  // staging mapping: thread -> (row, 32B segment)
  const int srow = tid >> 4;             // 0..15
  const int sseg = tid & 15;             // 0..15  (16 halves each)

  uint4  sq0, sq1;                       // bf16 staging regs (8 VGPRs)
  float4 sf0, sf1, sf2, sf3;             // fp32 staging regs (fallback)

  auto stageLoad = [&](int t) {
    if (BF16WX) {
      const u16* src = wxbf + ((size_t)(b0 + srow) * TT + t) * NN + sseg * 16;
      sq0 = *(const uint4*)(src);
      sq1 = *(const uint4*)(src + 8);
    } else {
      const float* src = wx + ((size_t)(b0 + srow) * TT + t) * NN + sseg * 16;
      sf0 = *(const float4*)(src);
      sf1 = *(const float4*)(src + 4);
      sf2 = *(const float4*)(src + 8);
      sf3 = *(const float4*)(src + 12);
    }
  };
  auto stageStore = [&](int buf) {
    u16* dst = &wxst[buf][srow * WXP + sseg * 16];
    if (BF16WX) {
      *(uint4*)(dst)     = sq0;
      *(uint4*)(dst + 8) = sq1;
    } else {
      dst[0]  = f2bf(sf0.x); dst[1]  = f2bf(sf0.y);
      dst[2]  = f2bf(sf0.z); dst[3]  = f2bf(sf0.w);
      dst[4]  = f2bf(sf1.x); dst[5]  = f2bf(sf1.y);
      dst[6]  = f2bf(sf1.z); dst[7]  = f2bf(sf1.w);
      dst[8]  = f2bf(sf2.x); dst[9]  = f2bf(sf2.y);
      dst[10] = f2bf(sf2.z); dst[11] = f2bf(sf2.w);
      dst[12] = f2bf(sf3.x); dst[13] = f2bf(sf3.y);
      dst[14] = f2bf(sf3.z); dst[15] = f2bf(sf3.w);
    }
  };

  // prologue: stage t=0
  stageLoad(0);
  stageStore(0);
  __syncthreads();

  const int row = tid >> 4;              // elementwise-phase mapping
  const int c0  = (tid & 15) * 8;

  for (int t = 0; t < TT; ++t) {
    const int cur = t & 1;

    // issue next step's wx loads NOW; the matching wait is at stageStore below
    if (t + 1 < TT) stageLoad(t + 1);

    // ---- GEMM: acc[j] = A(16x32 chunks) x B(32x16) accumulated over K ----
    v8f acc[4];
#pragma unroll
    for (int j = 0; j < 4; ++j)
#pragma unroll
      for (int r = 0; r < 8; ++r) acc[j][r] = 0.0f;

#pragma unroll
    for (int c = 0; c < KCH; ++c) {
      const u16* pa = (c < 8) ? &wxst[cur][lm * WXP + c * 32 + hi * 8]
                              : &h_bf[lm][(c - 8) * 32 + hi * 8];
      V16 a;
      a.q[0] = *(const uint4*)(pa);
      a.q[1] = *(const uint4*)(pa + 16);
#pragma unroll
      for (int j = 0; j < 4; ++j) {
        const int ncol = ((wave << 2) + j) * 16 + lm;   // B column == Wcat row
        const u16* q = Wcat + (size_t)ncol * KK + c * 32 + hi * 8;
        V16 bb;
        bb.q[0] = *(const uint4*)(q);
        bb.q[1] = *(const uint4*)(q + 16);
        acc[j] = __builtin_amdgcn_wmma_f32_16x16x32_bf16(
            /*neg_a=*/false, a.v, /*neg_b=*/false, bb.v,
            /*c_mod=*/(short)0, acc[j], /*reuse_a=*/false, /*reuse_b=*/false);
      }
    }

    // ---- spill gates to LDS (D layout: row = r + 8*hi, col = tile + lm) ----
#pragma unroll
    for (int j = 0; j < 4; ++j) {
      const int n0 = ((wave << 2) + j) * 16 + lm;
#pragma unroll
      for (int r = 0; r < 8; ++r) gates[r + (hi << 3)][n0] = acc[j][r];
    }
    __syncthreads();

    // ---- fused LSTM cell: 8 consecutive h-cols per thread ----
    float hrow[8];
#pragma unroll
    for (int k = 0; k < 8; ++k) {
      const int col = c0 + k;
      float ig = gates[row][col]           + bg[col];
      float fg = gates[row][col + HH]      + bg[col + HH];
      float gg = gates[row][col + 2 * HH]  + bg[col + 2 * HH];
      float og = gates[row][col + 3 * HH]  + bg[col + 3 * HH];
      float cp = c_st[row][col];
      float si = 1.0f / (1.0f + __expf(-ig));
      float sf = 1.0f / (1.0f + __expf(-fg));
      float so = 1.0f / (1.0f + __expf(-og));
      float c2 = sf * cp + si * tanhf(gg);
      float h2 = so * tanhf(c2);
      c_st[row][col] = c2;
      h_bf[row][col] = f2bf(h2);
      hrow[k] = h2;
    }
    float* ep = enc + ((size_t)(b0 + row) * TT + t) * HH + c0;
    *(float4*)(ep)     = make_float4(hrow[0], hrow[1], hrow[2], hrow[3]);
    *(float4*)(ep + 4) = make_float4(hrow[4], hrow[5], hrow[6], hrow[7]);

    // park next step's wx tile in the other LDS buffer (loadcnt wait lands here)
    if (t + 1 < TT) stageStore(cur ^ 1);
    __syncthreads();                      // state + staged tile ready for t+1
  }
}

// ---------------------------------------------------------------------------
extern "C" void kernel_launch(void* const* d_in, const int* in_sizes, int n_in,
                              void* d_out, int out_size, void* d_ws, size_t ws_size,
                              hipStream_t stream) {
  const int*   uin  = (const int*)d_in[0];
  const int*   sin  = (const int*)d_in[1];
  const float* ctx  = (const float*)d_in[2];
  const float* iv   = (const float*)d_in[3];
  const float* utab = (const float*)d_in[4];
  const float* stab = (const float*)d_in[5];
  const float* attw = (const float*)d_in[6];
  // d_in[7] = attn_b: cancels exactly under softmax shift-invariance
  const float* Wih  = (const float*)d_in[8];
  const float* Whh  = (const float*)d_in[9];
  const float* bih  = (const float*)d_in[10];
  const float* bhh  = (const float*)d_in[11];

  float* out  = (float*)d_out;
  float* wxo  = out;                                   // (B,T,N) input_weighted
  float* enc  = out + (size_t)BB * TT * NN;            // (B,T,H) input_encoded

  const size_t wcat_bytes = (size_t)GG * KK * sizeof(u16);       // 384 KB
  const size_t wxbf_bytes = (size_t)BB * TT * NN * sizeof(u16);  // 128 MB
  u16* Wcat = (u16*)d_ws;
  u16* wxbf = (u16*)((char*)d_ws + wcat_bytes);
  const bool bf_path = (ws_size >= wcat_bytes + wxbf_bytes);     // deterministic

  pack_w_kernel<<<(GG * KK + 255) / 256, 256, 0, stream>>>(Wih, Whh, Wcat);
  if (bf_path) {
    attn_wx_kernel<true><<<BB, 256, 0, stream>>>(uin, sin, ctx, iv, utab, stab,
                                                 attw, wxo, wxbf);
    lstm_rec_kernel<true><<<BB / ROWS, 256, 0, stream>>>(wxo, wxbf, Wcat,
                                                         bih, bhh, enc);
  } else {
    attn_wx_kernel<false><<<BB, 256, 0, stream>>>(uin, sin, ctx, iv, utab, stab,
                                                  attw, wxo, nullptr);
    lstm_rec_kernel<false><<<BB / ROWS, 256, 0, stream>>>(wxo, nullptr, Wcat,
                                                          bih, bhh, enc);
  }
}
